// CRFSegmentationModel_36155034697910
// MI455X (gfx1250) — matrix operand order repaired and checked by reading, not compiled, not run on hardware
//
#include <hip/hip_runtime.h>
#include <hip/hip_bf16.h>
#include <stdint.h>

// ---------------- problem constants ----------------
#define BATCH   16
#define CIN     3
#define HDIM    128
#define WDIM    128
#define LSEQ    (HDIM * WDIM)      // 16384
#define HID     256
#define NC      21
#define NCPAD   32
#define K1PAD   32                  // 27 -> 32
#define MTOT    (BATCH * LSEQ)      // 262144 spatial rows
#define SCH     128                 // viterbi chunk length
#define GCH     (LSEQ / SCH)        // 128 chunks

typedef __attribute__((ext_vector_type(16))) _Float16 v16h;
typedef __attribute__((ext_vector_type(8)))  _Float16 v8h;
typedef __attribute__((ext_vector_type(8)))  float    v8f;
typedef __attribute__((ext_vector_type(4)))  unsigned int u32x4;
typedef __attribute__((ext_vector_type(8)))  int i32x8;
typedef __attribute__((ext_vector_type(4)))  int i32x4;
typedef __attribute__((ext_vector_type(4)))  int v4i;

// ------------- CDNA5 feature gates (compile-safe) -------------
#if __has_builtin(__builtin_amdgcn_global_load_async_to_lds_b128) && \
    __has_builtin(__builtin_amdgcn_global_store_async_from_lds_b128) && \
    __has_builtin(__builtin_amdgcn_s_wait_asynccnt)
#define HAS_ASYNC 1
#else
#define HAS_ASYNC 0
#endif

#if __has_builtin(__builtin_amdgcn_tensor_load_to_lds) && \
    __has_builtin(__builtin_amdgcn_s_wait_tensorcnt)
#define HAS_TDM 1
#else
#define HAS_TDM 0
#endif

// typed address-space pointers for async b128 builtins
typedef __attribute__((address_space(1))) v4i gv4i;
typedef __attribute__((address_space(3))) v4i lv4i;
static __device__ __forceinline__ gv4i* as_gbl4(const void* p) {
    return (gv4i*)(uintptr_t)p;
}
static __device__ __forceinline__ lv4i* as_lds4(const void* p) {
    return (lv4i*)(unsigned)(uintptr_t)p;
}

__device__ __forceinline__ v8f wmma_f16(v16h a, v16h b, v8f c) {
    return __builtin_amdgcn_wmma_f32_16x16x32_f16(false, a, false, b,
                                                  (short)0, c, false, false);
}

__device__ __forceinline__ v16h load2x8(const _Float16* p0, const _Float16* p1) {
    union { struct { v8h l, h; } s; v16h v; } u;
    u.s.l = *(const v8h*)p0;
    u.s.h = *(const v8h*)p1;
    return u.v;
}

// Stage `tileRows` x 21 f32 emission rows into LDS via TDM (OOB rows -> 0),
// fallback: plain loads. Call from one wave; follow with __syncthreads().
__device__ __forceinline__ void stage_rows(const float* gsrc, float* dst,
                                           int tileRows, int availRows,
                                           int tid, int nthreads)
{
#if HAS_TDM
    (void)tid; (void)nthreads;
    unsigned long long ga = (unsigned long long)(uintptr_t)gsrc;
    u32x4 G0 = { 1u,                                      // count=1
                 (unsigned)(uintptr_t)(void*)dst,         // lds_addr
                 (unsigned)ga,                            // global_addr lo
                 (unsigned)((ga >> 32) & 0x1FFFFFFu) | (2u << 30) };
    unsigned d1 = (unsigned)availRows;                    // tensor_dim1 (OOB->0)
    i32x8 G1 = { (int)(2u << 16),                                  // data_size=4B
                 (int)(((unsigned)NC & 0xFFFFu) << 16),            // dim0 lo16
                 (int)((d1 & 0xFFFFu) << 16),                      // dim0 hi | dim1 lo
                 (int)((d1 >> 16) | ((unsigned)NC << 16)),         // dim1 hi | tile0
                 (int)(unsigned)tileRows,                          // tile1 | tile2=0
                 (int)(unsigned)NC,                                // dim0 stride lo32
                 0, 0 };
    i32x4 GZ = { 0, 0, 0, 0 };
#if __clang_major__ >= 23
    i32x8 GZ8 = { 0, 0, 0, 0, 0, 0, 0, 0 };
    __builtin_amdgcn_tensor_load_to_lds(G0, G1, GZ, GZ, GZ8, 0);
#else
    __builtin_amdgcn_tensor_load_to_lds(G0, G1, GZ, GZ, 0);
#endif
    __builtin_amdgcn_s_wait_tensorcnt(0);
#else
    int tot = tileRows * NC, avail = availRows * NC;
    for (int i = tid; i < tot; i += nthreads)
        dst[i] = (i < avail) ? gsrc[i] : 0.0f;
#endif
}

// =====================================================================
// Kernel A: fused conv1(3x3,3->256)+bias+ReLU -> conv2(1x1,256->21)+bias
// =====================================================================
__global__ void __launch_bounds__(64)
crf_backbone_wmma(const float* __restrict__ x,
                  const float* __restrict__ w1,
                  const float* __restrict__ b1,
                  const float* __restrict__ w2,
                  const float* __restrict__ b2,
                  float* __restrict__ emis)
{
    __shared__ _Float16 w1T[HID][K1PAD];
    __shared__ _Float16 w2T[NCPAD][HID];
    __shared__ float    b1s[HID];
    __shared__ float    b2s[NCPAD];
    __shared__ _Float16 Alds[2][16][K1PAD];
    __shared__ _Float16 Hlds[2][16][HID];
    __shared__ __align__(16) float Elds[2][16 * NC];

    const int tid  = threadIdx.x;
    const int wv   = tid >> 5;
    const int lane = tid & 31;
    const int col  = lane & 15;
    const int kb   = (lane < 16) ? 0 : 8;
    const int ob   = (lane < 16) ? 0 : 16;
    const int mr0  = (lane < 16) ? 0 : 8;

    for (int idx = tid; idx < HID * K1PAD; idx += 64) {
        int n = idx >> 5, k = idx & 31;
        float v = (k < 27) ? w1[n * 27 + k] : 0.0f;
        w1T[n][k] = (_Float16)v;
    }
    for (int idx = tid; idx < NCPAD * HID; idx += 64) {
        int n = idx >> 8, k = idx & 255;
        float v = (n < NC) ? w2[n * HID + k] : 0.0f;
        w2T[n][k] = (_Float16)v;
    }
    for (int i = tid; i < HID; i += 64) b1s[i] = b1[i];
    if (tid < NCPAD) b2s[tid] = (tid < NC) ? b2[tid] : 0.0f;
    __syncthreads();

    for (int it = 0; it < 8; ++it) {
        const int tile = blockIdx.x * 16 + wv * 8 + it;

        for (int idx = lane; idx < 16 * K1PAD; idx += 32) {
            int m = idx >> 5, k = idx & 31;
            float v = 0.0f;
            if (k < 27) {
                int p   = tile * 16 + m;
                int b   = p >> 14;
                int pos = p & (LSEQ - 1);
                int yy0 = pos >> 7, xx0 = pos & 127;
                int c   = k / 9;
                int rem = k - c * 9;
                int dy  = rem / 3;
                int dx  = rem - dy * 3;
                int yy  = yy0 + dy - 1, xs = xx0 + dx - 1;
                if ((unsigned)yy < (unsigned)HDIM && (unsigned)xs < (unsigned)WDIM)
                    v = x[((b * CIN + c) << 14) + (yy << 7) + xs];
            }
            Alds[wv][m][k] = (_Float16)v;
        }
        __builtin_amdgcn_wave_barrier();

        const _Float16* arow = &Alds[wv][col][0];
        v16h afrag = load2x8(arow + ob, arow + ob + 8);
        for (int n = 0; n < 16; ++n) {
            const _Float16* brow = &w1T[n * 16 + col][0];
            v16h bfrag = load2x8(brow + ob, brow + ob + 8);
            v8f  zero  = {};
            v8f  acc   = wmma_f16(afrag, bfrag, zero);
            float bias = b1s[n * 16 + col];
#pragma unroll
            for (int r = 0; r < 8; ++r) {
                float v = acc[r] + bias;
                Hlds[wv][mr0 + r][n * 16 + col] = (_Float16)fmaxf(v, 0.0f);
            }
        }
        __builtin_amdgcn_wave_barrier();

        v8f acc0 = {}, acc1 = {};
        const _Float16* hrow = &Hlds[wv][col][0];
#pragma unroll
        for (int kk = 0; kk < 8; ++kk) {
            v16h a2 = load2x8(hrow + kk * 32 + ob, hrow + kk * 32 + ob + 8);
            const _Float16* b0r = &w2T[col][kk * 32];
            const _Float16* b1r = &w2T[16 + col][kk * 32];
            v16h bf0 = load2x8(b0r + ob, b0r + ob + 8);
            v16h bf1 = load2x8(b1r + ob, b1r + ob + 8);
            acc0 = wmma_f16(a2, bf0, acc0);
            acc1 = wmma_f16(a2, bf1, acc1);
        }

        float bz0 = b2s[col];
        float bz1 = b2s[16 + col];
#pragma unroll
        for (int r = 0; r < 8; ++r) {
            int m = mr0 + r;
            Elds[wv][m * NC + col] = acc0[r] + bz0;
            if (col < NC - 16) Elds[wv][m * NC + 16 + col] = acc1[r] + bz1;
        }
#if HAS_ASYNC
#if __has_builtin(__builtin_amdgcn_s_wait_dscnt)
        __builtin_amdgcn_s_wait_dscnt(0);
#else
        __syncthreads();
#endif
        for (int i = lane; i < (16 * NC) / 4; i += 32)
            __builtin_amdgcn_global_store_async_from_lds_b128(
                as_gbl4(emis + (size_t)tile * (16 * NC) + i * 4),
                as_lds4(&Elds[wv][i * 4]), 0, 0);
        __builtin_amdgcn_s_wait_asynccnt(0);
#else
        __builtin_amdgcn_wave_barrier();
        for (int i = lane; i < 16 * NC; i += 32)
            emis[(size_t)tile * (16 * NC) + i] = Elds[wv][i];
#endif
        __builtin_amdgcn_wave_barrier();
    }
}

// =====================================================================
// Phase 1: per (batch, chunk) compose the 21x21 max-plus transfer matrix
// over SCH steps. Thread = (i, n) output element. Emissions via TDM.
// =====================================================================
__global__ void __launch_bounds__(448)
crf_chunk_compose(const float* __restrict__ emis,
                  const float* __restrict__ trans,
                  float* __restrict__ Mg)          // [B*G, 21, 21]
{
    __shared__ float Cb[2][NC][NC];
    __shared__ __align__(16) float ebuf[SCH * NC];

    const int tid = threadIdx.x;
    const int bg  = blockIdx.x;
    const int b   = bg >> 7, g = bg & (GCH - 1);
    const bool act = tid < NC * NC;
    const int i = act ? (tid / NC) : 0;
    const int n = act ? (tid - (tid / NC) * NC) : 0;

    const float* eb = emis + ((size_t)b * LSEQ + (size_t)g * SCH + 1) * NC;
    if (tid < 32) stage_rows(eb, ebuf, SCH, LSEQ - 1 - g * SCH, tid, 32);

    float trn[NC];
#pragma unroll
    for (int p = 0; p < NC; ++p) trn[p] = act ? trans[p * NC + n] : 0.0f;
    if (act) Cb[0][i][n] = (i == n) ? 0.0f : -1.0e30f;   // max-plus identity
    __syncthreads();

    int rem = LSEQ - 1 - g * SCH;
    const int nst = (rem < SCH) ? rem : SCH;
    int cur = 0;
    for (int s = 0; s < nst; ++s) {
        if (act) {
            float nv = -1.0e30f;
#pragma unroll
            for (int p = 0; p < NC; ++p)
                nv = fmaxf(nv, Cb[cur][i][p] + trn[p]);
            Cb[cur ^ 1][i][n] = nv + ebuf[s * NC + n];
        }
        __syncthreads();
        cur ^= 1;
    }
    if (act) Mg[((size_t)bg * NC + i) * NC + n] = Cb[cur][i][n];
}

// =====================================================================
// Phase 2: serial scan over chunk matrices (per batch). Writes chunk
// boundary scores SB[b][g][21] and last_tag[b].
// =====================================================================
__global__ void __launch_bounds__(32)
crf_chunk_scan(const float* __restrict__ emis,
               const float* __restrict__ start,
               const float* __restrict__ endt,
               const float* __restrict__ Mg,
               float* __restrict__ SB,             // [B, G, 21]
               int* __restrict__ last_tag)
{
    __shared__ float sb[32];
    const int b = blockIdx.x, lane = threadIdx.x;
    const bool act = lane < NC;
    const int n = act ? lane : 0;

    float s = act ? (emis[(size_t)b * LSEQ * NC + lane] + start[lane]) : -1.0e30f;
    sb[lane] = s;
    __builtin_amdgcn_wave_barrier();

    for (int g = 0; g < GCH; ++g) {
        if (act) SB[((size_t)b * GCH + g) * NC + lane] = s;
        const float* M = Mg + (size_t)(b * GCH + g) * NC * NC;
        float nv = -1.0e30f;
#pragma unroll
        for (int i2 = 0; i2 < NC; ++i2)
            nv = fmaxf(nv, sb[i2] + M[i2 * NC + n]);
        __builtin_amdgcn_wave_barrier();   // all sb reads done
        s = act ? nv : -1.0e30f;
        sb[lane] = s;
        __builtin_amdgcn_wave_barrier();
    }

    sb[lane] = act ? (s + endt[lane]) : -3.0e38f;
    __builtin_amdgcn_wave_barrier();
    if (lane == 0) {
        float best = -3.4e38f; int bt = 0;
#pragma unroll
        for (int k = 0; k < NC; ++k) {
            float v = sb[k];
            if (v > best) { best = v; bt = k; }
        }
        last_tag[b] = bt;
    }
}

// =====================================================================
// Phase 3: per (batch, chunk) replay SCH steps from the true boundary
// score, writing backpointers. Emissions via TDM.
// =====================================================================
__global__ void __launch_bounds__(32)
crf_chunk_replay(const float* __restrict__ emis,
                 const float* __restrict__ trans,
                 const float* __restrict__ SB,
                 unsigned char* __restrict__ hist) // [(L-1), B, 21]
{
    __shared__ float sc[2][32];
    __shared__ __align__(16) float ebuf[SCH * NC];

    const int bg = blockIdx.x;
    const int b = bg >> 7, g = bg & (GCH - 1);
    const int lane = threadIdx.x;
    const bool act = lane < NC;
    const int nl = act ? lane : 0;

    const float* eb = emis + ((size_t)b * LSEQ + (size_t)g * SCH + 1) * NC;
    stage_rows(eb, ebuf, SCH, LSEQ - 1 - g * SCH, lane, 32);

    float tr[NC];
#pragma unroll
    for (int p = 0; p < NC; ++p) tr[p] = act ? trans[p * NC + lane] : 0.0f;

    sc[0][lane] = act ? SB[((size_t)b * GCH + g) * NC + lane] : -3.0e38f;
    __syncthreads();

    int rem = LSEQ - 1 - g * SCH;
    const int nst = (rem < SCH) ? rem : SCH;
    const int t0 = g * SCH;
    int cur = 0;
    for (int s = 0; s < nst; ++s) {
        float e = ebuf[s * NC + nl];
        float best = -3.0e38f; int bp = 0;
#pragma unroll
        for (int p = 0; p < NC; ++p) {
            float c = sc[cur][p] + tr[p];
            if (c > best) { best = c; bp = p; }
        }
        if (act)
            hist[(size_t)(t0 + s) * (BATCH * NC) + b * NC + lane] =
                (unsigned char)bp;
        sc[cur ^ 1][lane] = act ? (best + e) : -3.0e38f;
        __builtin_amdgcn_wave_barrier();
        cur ^= 1;
    }
}

// =====================================================================
// Phase 4: per (batch, chunk), 21 lanes walk the hist chain for every
// possible entry tag, recording chainTag[t][b][j].
// =====================================================================
__global__ void __launch_bounds__(32)
crf_chain_maps(const unsigned char* __restrict__ hist,
               unsigned char* __restrict__ chainTag)
{
    const int bg = blockIdx.x;
    const int b = bg >> 7, g = bg & (GCH - 1);
    const int j = threadIdx.x;
    if (j >= NC) return;
    const int bot = g * SCH;
    const int top = (g == GCH - 1) ? (LSEQ - 1) : (g + 1) * SCH;
    int f = j;
    for (int t = top - 1; t >= bot; --t) {
        f = hist[(size_t)t * (BATCH * NC) + b * NC + f];
        chainTag[(size_t)t * (BATCH * NC) + b * NC + j] = (unsigned char)f;
    }
}

// =====================================================================
// Phase 5: serial boundary-tag selection (128 LUT steps per batch).
// =====================================================================
__global__ void __launch_bounds__(32)
crf_chain_select(const unsigned char* __restrict__ chainTag,
                 const int* __restrict__ last_tag,
                 int* __restrict__ entrySel)       // [B, G]
{
    const int b = threadIdx.x;
    if (b >= BATCH) return;
    int tag = last_tag[b];
    for (int g = GCH - 1; g >= 0; --g) {
        entrySel[b * GCH + g] = tag;
        tag = chainTag[(size_t)(g * SCH) * (BATCH * NC) + b * NC + tag];
    }
}

// =====================================================================
// Phase 6: fully parallel tag gather into the output.
// =====================================================================
__global__ void __launch_bounds__(256)
crf_emit_tags(const unsigned char* __restrict__ chainTag,
              const int* __restrict__ entrySel,
              const int* __restrict__ last_tag,
              int* __restrict__ out)               // [B, H, W] int32
{
    int idx = blockIdx.x * 256 + threadIdx.x;
    if (idx >= MTOT) return;
    int b = idx >> 14, t = idx & (LSEQ - 1);
    int v;
    if (t == LSEQ - 1) {
        v = last_tag[b];
    } else {
        int g = t >> 7;                    // SCH == 128
        int sel = entrySel[b * GCH + g];
        v = chainTag[(size_t)t * (BATCH * NC) + b * NC + sel];
    }
    out[idx] = v;
}

// =====================================================================
extern "C" void kernel_launch(void* const* d_in, const int* in_sizes, int n_in,
                              void* d_out, int out_size, void* d_ws, size_t ws_size,
                              hipStream_t stream) {
    (void)in_sizes; (void)n_in; (void)out_size; (void)ws_size;

    const float* x      = (const float*)d_in[0];
    const float* w1     = (const float*)d_in[1];
    const float* b1     = (const float*)d_in[2];
    const float* w2     = (const float*)d_in[3];
    const float* b2     = (const float*)d_in[4];
    const float* start  = (const float*)d_in[5];
    const float* endt   = (const float*)d_in[6];
    const float* trans  = (const float*)d_in[7];

    // ---- workspace layout (bytes) ----
    char* ws = (char*)d_ws;
    float*         emis     = (float*)(ws);                       // 22,020,096
    unsigned char* hist     = (unsigned char*)(ws + 22020096);    //  5,505,024
    unsigned char* chainTag = (unsigned char*)(ws + 27525120);    //  5,505,024
    float*         Mg       = (float*)(ws + 33030144);            //  3,612,672
    float*         SB       = (float*)(ws + 36642816);            //    172,032
    int*           entrySel = (int*)(ws + 36814848);              //      8,192
    int*           last_tag = (int*)(ws + 36823040);              //         64

    int* out = (int*)d_out;

    crf_backbone_wmma<<<dim3(1024), dim3(64), 0, stream>>>(
        x, w1, b1, w2, b2, emis);

    crf_chunk_compose<<<dim3(BATCH * GCH), dim3(448), 0, stream>>>(
        emis, trans, Mg);

    crf_chunk_scan<<<dim3(BATCH), dim3(32), 0, stream>>>(
        emis, start, endt, Mg, SB, last_tag);

    crf_chunk_replay<<<dim3(BATCH * GCH), dim3(32), 0, stream>>>(
        emis, trans, SB, hist);

    crf_chain_maps<<<dim3(BATCH * GCH), dim3(32), 0, stream>>>(
        hist, chainTag);

    crf_chain_select<<<dim3(1), dim3(32), 0, stream>>>(
        chainTag, last_tag, entrySel);

    crf_emit_tags<<<dim3(MTOT / 256), dim3(256), 0, stream>>>(
        chainTag, entrySel, last_tag, out);
}
